// LlamaAttention_49538152792934
// MI455X (gfx1250) — matrix operand order repaired and checked by reading, not compiled, not run on hardware
//
#include <hip/hip_runtime.h>

#define DI __device__ __forceinline__

typedef __attribute__((ext_vector_type(16))) __bf16 v16bf;
typedef __attribute__((ext_vector_type(8)))  __bf16 v8bf;
typedef __attribute__((ext_vector_type(4)))  __bf16 v4bf;
typedef __attribute__((ext_vector_type(8)))  float  v8f;
typedef unsigned int u32x4 __attribute__((ext_vector_type(4)));
typedef int          i32x8 __attribute__((ext_vector_type(8)));
typedef int          i32x4 __attribute__((ext_vector_type(4)));

constexpr int S  = 2048;
constexpr int D  = 4096;
constexpr int H  = 32;
constexpr int KVH = 8;
constexpr int HD = 128;
constexpr int G  = H / KVH;
constexpr int NQ = H * HD;    // 4096
constexpr int NK = KVH * HD;  // 1024
constexpr float ATTN_SCALE = 0.08838834764831845f;  // 1/sqrt(128)
constexpr float LN1E4 = 9.210340371976184f;         // ln(10000)

#if __has_builtin(__builtin_amdgcn_tensor_load_to_lds)
#define HAVE_TDM 1
#endif

DI unsigned short f2bf_bits(float f) {
  unsigned u = __builtin_bit_cast(unsigned, f);
  unsigned r = u + 0x7FFFu + ((u >> 16) & 1u);  // round-to-nearest-even
  return (unsigned short)(r >> 16);
}
DI __bf16 f2bf(float f) { return __builtin_bit_cast(__bf16, f2bf_bits(f)); }
DI unsigned pack2bf(float a, float b) {
  return (unsigned)f2bf_bits(a) | ((unsigned)f2bf_bits(b) << 16);
}

DI v16bf mk16(v8bf lo, v8bf hi) {
  v16bf r;
#pragma unroll
  for (int i = 0; i < 8; ++i) { r[i] = lo[i]; r[i + 8] = hi[i]; }
  return r;
}

#ifdef HAVE_TDM
// 2D bf16 tile DMA: global (row-major, elements) -> LDS, per CDNA5 D# layout.
DI void tdm_load_2d_bf16(unsigned lds_off, const void* gaddr,
                         unsigned tile_d0, unsigned tile_d1,
                         unsigned tensor_d0, unsigned tensor_d1,
                         unsigned stride0) {
  unsigned long long ga = (unsigned long long)gaddr;
  u32x4 g0;
  g0[0] = 1u;                                  // count=1, user mode
  g0[1] = lds_off;                             // lds_addr (bytes)
  g0[2] = (unsigned)ga;                        // global_addr[31:0]
  g0[3] = (unsigned)((ga >> 32) & 0x01FFFFFFu) // global_addr[56:32]
          | (2u << 30);                        // type=2 ("image")
  i32x8 g1;
  g1[0] = (int)(1u << 16);                     // workgroup_mask=0, data_size=1 (2B)
  g1[1] = (int)(tensor_d0 << 16);              // tensor_dim0[15:0] @ bits 63:48
  g1[2] = (int)((tensor_d0 >> 16) | (tensor_d1 << 16));  // dim0 hi | dim1 lo
  g1[3] = (int)((tensor_d1 >> 16) | (tile_d0 << 16));    // dim1 hi | tile_dim0
  g1[4] = (int)(tile_d1 & 0xFFFFu);            // tile_dim1 | tile_dim2=0
  g1[5] = (int)stride0;                        // tensor_dim0_stride[31:0]
  g1[6] = 0;
  g1[7] = 0;
  i32x4 z4 = {0, 0, 0, 0};
#if __clang_major__ >= 23
  i32x8 z8 = {0, 0, 0, 0, 0, 0, 0, 0};
  __builtin_amdgcn_tensor_load_to_lds(g0, g1, z4, z4, z8, 0);
#else
  __builtin_amdgcn_tensor_load_to_lds(g0, g1, z4, z4, 0);
#endif
}
#endif

// ---------------------------------------------------------------- cvt f32->bf16 (vectorized)
__global__ __launch_bounds__(256)
void cvt_bf16_kernel(const float4* __restrict__ src, v4bf* __restrict__ dst, int n4) {
  int i = blockIdx.x * blockDim.x + threadIdx.x;
  int stride = gridDim.x * blockDim.x;
  for (; i < n4; i += stride) {
    float4 f = src[i];
    v4bf o;
    o[0] = f2bf(f.x); o[1] = f2bf(f.y); o[2] = f2bf(f.z); o[3] = f2bf(f.w);
    dst[i] = o;
  }
}

// ======================= shared GEMM tile machinery (128x256 block, K-tile 64, 2x buffered)
struct SMGemm {
  __bf16 A[2][128][64];   // 32KB  [m][k]
  __bf16 B[2][256][64];   // 64KB  [n][k] (transposed)
};

// stage one 64-deep K slab into buffer `buf`
DI void gemm_stage(SMGemm& g, int buf, int tid,
                   const __bf16* __restrict__ Abase, int lda,  // A row base (tile row 0), row-major
                   const __bf16* __restrict__ W, int ldw,      // B col-slice base, row-major [k][n]
                   int k0) {
  const int arow = tid >> 1, akc = (tid & 1) * 32;
  const __bf16* ga = Abase + (size_t)arow * lda + k0 + akc;
#pragma unroll
  for (int v = 0; v < 4; ++v)
    *(v8bf*)&g.A[buf][arow][akc + v * 8] = *(const v8bf*)(ga + v * 8);

  // B: 64k x 256n slab; each thread: 8 rows x 8 cols register transpose -> b128 stores
  const int krB = (tid >> 5) * 8;      // 0..56
  const int nbB = (tid & 31) * 8;      // 0..248
  const __bf16* gb = W + (size_t)(k0 + krB) * ldw + nbB;
  v8bf rowv[8];
#pragma unroll
  for (int r = 0; r < 8; ++r) rowv[r] = *(const v8bf*)(gb + (size_t)r * ldw);
#pragma unroll
  for (int j = 0; j < 8; ++j) {
    v8bf c;
#pragma unroll
    for (int r = 0; r < 8; ++r) c[r] = rowv[r][j];
    *(v8bf*)&g.B[buf][nbB + j][krB] = c;
  }
}

// one 64-deep compute stage: 32 wmma per wave (4x4 subtiles x 2 K-steps)
DI void gemm_compute(const SMGemm& g, int buf, int wm0, int wn0, int l15, int hi16,
                     v8f acc[4][4]) {
  const int akb = hi16 ? 8 : 0;
  const int bkb = hi16 ? 8 : 0;
#pragma unroll
  for (int ks = 0; ks < 64; ks += 32) {
    v16bf a[4], b[4];
#pragma unroll
    for (int i = 0; i < 4; ++i) {
      const __bf16* p = &g.A[buf][wm0 + i * 16 + l15][ks + akb];
      a[i] = mk16(*(const v8bf*)p, *(const v8bf*)(p + 16));
    }
#pragma unroll
    for (int j = 0; j < 4; ++j) {
      const __bf16* p = &g.B[buf][wn0 + j * 16 + l15][ks + bkb];
      b[j] = mk16(*(const v8bf*)p, *(const v8bf*)(p + 16));
    }
#pragma unroll
    for (int i = 0; i < 4; ++i)
#pragma unroll
      for (int j = 0; j < 4; ++j)
        acc[i][j] = __builtin_amdgcn_wmma_f32_16x16x32_bf16(
            false, a[i], false, b[j], (short)0, acc[i][j], false, false);
  }
}

// ---------------------------------------------------------------- fused QKV GEMM + bias + RoPE
// grid: (S/128, (NQ+2*NK)/256) ; block 256 (8 waves)
__global__ __launch_bounds__(256)
void qkv_rope_kernel(const __bf16* __restrict__ xb,
                     const __bf16* __restrict__ wqb,
                     const __bf16* __restrict__ wkb,
                     const __bf16* __restrict__ wvb,
                     const float*  __restrict__ bq,
                     const float*  __restrict__ bk,
                     const float*  __restrict__ bv,
                     __bf16* __restrict__ qout,   // [H][S][HD]
                     __bf16* __restrict__ kout,   // [KVH][S][HD]
                     __bf16* __restrict__ vout)   // [KVH][S][HD]
{
  __shared__ union {
    SMGemm g;               // 96KB, live during K loop
    __bf16 C[128][256];     // 64KB, live after K loop
  } sm;

  const int m0    = blockIdx.x * 128;
  const int ncol0 = blockIdx.y * 256;

  const __bf16* W; const float* bias; __bf16* base_out; int ldw, mode, head0;
  if (ncol0 < NQ) {
    W = wqb + ncol0; ldw = NQ; bias = bq + ncol0;
    base_out = qout; head0 = ncol0 / HD; mode = 0;
  } else if (ncol0 < NQ + NK) {
    int c = ncol0 - NQ;
    W = wkb + c; ldw = NK; bias = bk + c;
    base_out = kout; head0 = c / HD; mode = 1;
  } else {
    int c = ncol0 - NQ - NK;
    W = wvb + c; ldw = NK; bias = bv + c;
    base_out = vout; head0 = c / HD; mode = 2;
  }

  const int tid  = threadIdx.x;
  const int lane = tid & 31;
  const int wave = tid >> 5;
  const int wm0  = (wave >> 2) * 64;  // 0 / 64
  const int wn0  = (wave & 3) * 64;   // 0..192
  const int l15  = lane & 15;
  const int hi16 = lane >> 4;

  const __bf16* Abase = xb + (size_t)m0 * D;

  v8f acc[4][4] = {};

  gemm_stage(sm.g, 0, tid, Abase, D, W, ldw, 0);
  __syncthreads();
  for (int k0 = 0; k0 < D; k0 += 64) {
    const int buf = (k0 >> 6) & 1;
    if (k0 + 64 < D) {
      gemm_stage(sm.g, buf ^ 1, tid, Abase, D, W, ldw, k0 + 64);
      __builtin_prefetch((const void*)(W + (size_t)(k0 + 128) * ldw), 0, 1);
    }
    gemm_compute(sm.g, buf, wm0, wn0, l15, hi16, acc);
    __syncthreads();
  }

  // C layout -> LDS with bias (overlays GEMM buffers; safe after final barrier)
#pragma unroll
  for (int i = 0; i < 4; ++i)
#pragma unroll
    for (int j = 0; j < 4; ++j)
#pragma unroll
      for (int r = 0; r < 8; ++r) {
        int row = wm0 + i * 16 + r + hi16 * 8;
        int col = wn0 + j * 16 + l15;
        sm.C[row][col] = f2bf(acc[i][j][r] + bias[col]);
      }
  __syncthreads();

  // epilogue: two heads per block; RoPE (Q/K) or copy (V)
  const int row = tid >> 1;
  const int hb  = tid & 1;
  const int pos = m0 + row;
  __bf16* drow = base_out + ((size_t)(head0 + hb) * S + pos) * HD;
  const __bf16* crow = &sm.C[row][hb * 128];
  if (mode == 2) {
#pragma unroll
    for (int v = 0; v < 16; ++v)
      *(v8bf*)(drow + v * 8) = *(const v8bf*)(crow + v * 8);
  } else {
#pragma unroll
    for (int d = 0; d < 64; d += 2) {
      float o[4];
#pragma unroll
      for (int e = 0; e < 2; ++e) {
        int dd = d + e;
        float inv = __expf((float)dd * (-2.0f / HD) * LN1E4);
        float ang = (float)pos * inv;
        float sn, cs;
        __sincosf(ang, &sn, &cs);
        float x1 = (float)crow[dd];
        float x2 = (float)crow[dd + 64];
        o[e]     = x1 * cs - x2 * sn;
        o[2 + e] = x2 * cs + x1 * sn;
      }
      *(unsigned*)(drow + d)      = pack2bf(o[0], o[1]);
      *(unsigned*)(drow + d + 64) = pack2bf(o[2], o[3]);
    }
  }
}

// ---------------------------------------------------------------- flash attention (causal, GQA)
// grid: (S/128, H) ; block 256 (8 waves, 16 query rows per wave)
__global__ __launch_bounds__(256)
void attn_kernel(const __bf16* __restrict__ qb,   // [H][S][HD]
                 const __bf16* __restrict__ kb,   // [KVH][S][HD]
                 const __bf16* __restrict__ vb,   // [KVH][S][HD]
                 __bf16* __restrict__ ab)         // [S][NQ]
{
  __shared__ __bf16 lK[128][128];      // [key][hd]
  __shared__ __bf16 lV[128][128];      // [hd][key]  (transposed)
  __shared__ __bf16 lP[8][16][136];    // per-wave probs, padded rows

  const int qtile = blockIdx.x;
  const int h     = blockIdx.y;
  const int kvh   = h / G;
  const int tid = threadIdx.x, lane = tid & 31, wave = tid >> 5;
  const int l15 = lane & 15, hi16 = lane >> 4;
  const int qrow0 = qtile * 128 + wave * 16;
  const int akb = hi16 ? 8 : 0;
  const int bkb = hi16 ? 8 : 0;

  // Q fragments (registers, A-layout straight from row-major global)
  const __bf16* qrow = qb + ((size_t)h * S + qrow0 + l15) * HD;
  v16bf qf[4];
#pragma unroll
  for (int c = 0; c < 4; ++c)
    qf[c] = mk16(*(const v8bf*)(qrow + c * 32 + akb),
                 *(const v8bf*)(qrow + c * 32 + akb + 16));

  v8f oacc[8] = {};
  float mrow[8], lrow[8];
#pragma unroll
  for (int r = 0; r < 8; ++r) { mrow[r] = -3.0e38f; lrow[r] = 0.0f; }

  const __bf16* Kbase = kb + (size_t)kvh * S * HD;
  const __bf16* Vbase = vb + (size_t)kvh * S * HD;
#ifdef HAVE_TDM
  const unsigned ldsK_off = (unsigned)(size_t)(void*)&lK[0][0];
#endif

  // V transpose staging indices: 8 keys x 8 dims per thread, b128 LDS stores
  const int vkg = tid >> 4;        // 0..15 -> key group
  const int vdg = tid & 15;        // 0..15 -> dim group
  const int vkey0 = vkg * 8;
  const int vd0   = vdg * 8;

  for (int kt = 0; kt <= qtile; ++kt) {
    const int kb0 = kt * 128;
#ifdef HAVE_TDM
    if (wave == 0)
      tdm_load_2d_bf16(ldsK_off, (const void*)(Kbase + (size_t)kb0 * HD),
                       /*tile*/ HD, 128, /*tensor*/ HD, 128, /*stride0*/ HD);
#else
    {
      const int key = tid >> 1, c0 = (tid & 1) * 64;
      const __bf16* gk = Kbase + (size_t)(kb0 + key) * HD + c0;
#pragma unroll
      for (int v = 0; v < 8; ++v)
        *(v8bf*)&lK[key][c0 + v * 8] = *(const v8bf*)(gk + v * 8);
    }
#endif
    // V block transposed: register 8x8 transpose, b128 stores
    {
      const __bf16* gv = Vbase + (size_t)(kb0 + vkey0) * HD + vd0;
      v8bf rowv[8];
#pragma unroll
      for (int r = 0; r < 8; ++r) rowv[r] = *(const v8bf*)(gv + (size_t)r * HD);
#pragma unroll
      for (int j = 0; j < 8; ++j) {
        v8bf c;
#pragma unroll
        for (int r = 0; r < 8; ++r) c[r] = rowv[r][j];
        *(v8bf*)&lV[vd0 + j][vkey0] = c;
      }
    }
#ifdef HAVE_TDM
    if (wave == 0) __builtin_amdgcn_s_wait_tensorcnt(0);
#endif
    __syncthreads();

    // S = Q @ K^T  (contraction over HD)
    v8f sacc[8] = {};
#pragma unroll
    for (int n = 0; n < 8; ++n) {
      const __bf16* kr = &lK[n * 16 + l15][0];
#pragma unroll
      for (int c = 0; c < 4; ++c) {
        v16bf bf = mk16(*(const v8bf*)(kr + c * 32 + bkb),
                        *(const v8bf*)(kr + c * 32 + bkb + 16));
        sacc[n] = __builtin_amdgcn_wmma_f32_16x16x32_bf16(
            false, qf[c], false, bf, (short)0, sacc[n], false, false);
      }
    }

    // scale + causal mask on the diagonal block
    const bool diag = (kt == qtile);
#pragma unroll
    for (int n = 0; n < 8; ++n) {
      int keyA = kb0 + n * 16 + l15;
#pragma unroll
      for (int r = 0; r < 8; ++r) {
        float s = sacc[n][r] * ATTN_SCALE;
        if (diag && keyA > (qrow0 + r + hi16 * 8)) s = -3.0e38f;
        sacc[n][r] = s;
      }
    }

    // online softmax (rows r / r+8 per lane half; reduce across 16 lanes)
#pragma unroll
    for (int r = 0; r < 8; ++r) {
      float mx = -3.0e38f;
#pragma unroll
      for (int n = 0; n < 8; ++n) mx = fmaxf(mx, sacc[n][r]);
#pragma unroll
      for (int off = 1; off < 16; off <<= 1) mx = fmaxf(mx, __shfl_xor(mx, off, 32));
      float mnew = fmaxf(mrow[r], mx);
      float corr = __expf(mrow[r] - mnew);
      float ps = 0.0f;
#pragma unroll
      for (int n = 0; n < 8; ++n) {
        float p = __expf(sacc[n][r] - mnew);
        sacc[n][r] = p;
        ps += p;
      }
#pragma unroll
      for (int off = 1; off < 16; off <<= 1) ps += __shfl_xor(ps, off, 32);
      lrow[r] = lrow[r] * corr + ps;
      mrow[r] = mnew;
#pragma unroll
      for (int n = 0; n < 8; ++n) oacc[n][r] *= corr;
    }

    // P (C-layout f32) -> per-wave LDS (row-major bf16) for A-layout reload
#pragma unroll
    for (int n = 0; n < 8; ++n)
#pragma unroll
      for (int r = 0; r < 8; ++r)
        lP[wave][r + hi16 * 8][n * 16 + l15] = f2bf(sacc[n][r]);

    // O += P @ V  (contraction over keys)
#pragma unroll
    for (int kk = 0; kk < 4; ++kk) {
      const __bf16* pr = &lP[wave][l15][kk * 32 + akb];
      v16bf ap = mk16(*(const v8bf*)pr, *(const v8bf*)(pr + 16));
#pragma unroll
      for (int n = 0; n < 8; ++n) {
        const __bf16* vr = &lV[n * 16 + l15][kk * 32 + bkb];
        v16bf bv_ = mk16(*(const v8bf*)vr, *(const v8bf*)(vr + 16));
        oacc[n] = __builtin_amdgcn_wmma_f32_16x16x32_bf16(
            false, ap, false, bv_, (short)0, oacc[n], false, false);
      }
    }
    __syncthreads();
  }

  // normalize and store into [S][NQ]
#pragma unroll
  for (int r = 0; r < 8; ++r) {
    float invl = 1.0f / lrow[r];
    int row = qrow0 + r + hi16 * 8;
    __bf16* orow = ab + (size_t)row * NQ + h * HD;
#pragma unroll
    for (int n = 0; n < 8; ++n)
      orow[n * 16 + l15] = f2bf(oacc[n][r] * invl);
  }
}

// ---------------------------------------------------------------- output projection (fp32 out)
// grid: (S/128, D/256) ; block 256
__global__ __launch_bounds__(256)
void oproj_kernel(const __bf16* __restrict__ ab,   // [S][NQ]
                  const __bf16* __restrict__ wob,  // [NQ][D]
                  const float*  __restrict__ bo,   // [D]
                  float* __restrict__ out)         // [S][D]
{
  __shared__ SMGemm sm;

  const int m0 = blockIdx.x * 128;
  const int n0 = blockIdx.y * 256;

  const int tid  = threadIdx.x;
  const int lane = tid & 31;
  const int wave = tid >> 5;
  const int wm0  = (wave >> 2) * 64;
  const int wn0  = (wave & 3) * 64;
  const int l15  = lane & 15;
  const int hi16 = lane >> 4;

  const __bf16* Abase = ab + (size_t)m0 * NQ;
  const __bf16* W = wob + n0;

  v8f acc[4][4] = {};

  gemm_stage(sm, 0, tid, Abase, NQ, W, D, 0);
  __syncthreads();
  for (int k0 = 0; k0 < NQ; k0 += 64) {
    const int buf = (k0 >> 6) & 1;
    if (k0 + 64 < NQ) {
      gemm_stage(sm, buf ^ 1, tid, Abase, NQ, W, D, k0 + 64);
      __builtin_prefetch((const void*)(W + (size_t)(k0 + 128) * D), 0, 1);
    }
    gemm_compute(sm, buf, wm0, wn0, l15, hi16, acc);
    __syncthreads();
  }

#pragma unroll
  for (int i = 0; i < 4; ++i)
#pragma unroll
    for (int j = 0; j < 4; ++j)
#pragma unroll
      for (int r = 0; r < 8; ++r) {
        int row = m0 + wm0 + i * 16 + r + hi16 * 8;
        int col = n0 + wn0 + j * 16 + l15;
        out[(size_t)row * D + col] = acc[i][j][r] + bo[col];
      }
}

// ---------------------------------------------------------------- launch
extern "C" void kernel_launch(void* const* d_in, const int* in_sizes, int n_in,
                              void* d_out, int out_size, void* d_ws, size_t ws_size,
                              hipStream_t stream) {
  const float* x  = (const float*)d_in[0];
  const float* wq = (const float*)d_in[1];
  const float* bq = (const float*)d_in[2];
  const float* wk = (const float*)d_in[3];
  const float* bk = (const float*)d_in[4];
  const float* wv = (const float*)d_in[5];
  const float* bv = (const float*)d_in[6];
  const float* wo = (const float*)d_in[7];
  const float* bo = (const float*)d_in[8];
  float* out = (float*)d_out;

  // workspace layout (bf16 elements)
  __bf16* p = (__bf16*)d_ws;
  __bf16* xb  = p; p += (size_t)S * D;
  __bf16* wqb = p; p += (size_t)D * NQ;
  __bf16* wkb = p; p += (size_t)D * NK;
  __bf16* wvb = p; p += (size_t)D * NK;
  __bf16* wob = p; p += (size_t)NQ * D;
  __bf16* qb  = p; p += (size_t)H * S * HD;
  __bf16* kbf = p; p += (size_t)KVH * S * HD;
  __bf16* vbf = p; p += (size_t)KVH * S * HD;
  __bf16* abf = p; p += (size_t)S * NQ;

  auto cvt = [&](const float* s, __bf16* d, size_t n) {
    int n4 = (int)(n / 4);
    int blocks = (n4 + 255) / 256;
    if (blocks > 4096) blocks = 4096;
    cvt_bf16_kernel<<<blocks, 256, 0, stream>>>((const float4*)s, (v4bf*)d, n4);
  };
  cvt(x,  xb,  (size_t)S * D);
  cvt(wq, wqb, (size_t)D * NQ);
  cvt(wk, wkb, (size_t)D * NK);
  cvt(wv, wvb, (size_t)D * NK);
  cvt(wo, wob, (size_t)NQ * D);

  qkv_rope_kernel<<<dim3(S / 128, (NQ + 2 * NK) / 256), 256, 0, stream>>>(
      xb, wqb, wkb, wvb, bq, bk, bv, qb, kbf, vbf);

  attn_kernel<<<dim3(S / 128, H), 256, 0, stream>>>(qb, kbf, vbf, abf);

  oproj_kernel<<<dim3(S / 128, D / 256), 256, 0, stream>>>(abf, wob, bo, out);
}
